// VictorASIFractalLightModel_30365418782769
// MI455X (gfx1250) — matrix operand order repaired and checked by reading, not compiled, not run on hardware
//
#include <hip/hip_runtime.h>
#include <hip/hip_bf16.h>
#include <hip/hip_fp16.h>

// ---------------------------------------------------------------------------
// MI455X (gfx1250): WMMA f16 (f32 accumulate) for all contractions.
// All GEMM operands kept in f16 buffers so every LDS tile is staged with
// async global->LDS b128 copies (ASYNCcnt), double-buffered to overlap
// copy of tile t+1 with WMMA on tile t.
// ---------------------------------------------------------------------------

typedef _Float16 half_t;
typedef __attribute__((ext_vector_type(16))) _Float16 v16h;
typedef __attribute__((ext_vector_type(8)))  float    v8f;

#define WMMA_F32_F16(a, b, c) \
  __builtin_amdgcn_wmma_f32_16x16x32_f16(false, (a), false, (b), (short)0, (c), false, false)

constexpr int Bb = 2, Ll = 2048, Dd = 1024;
constexpr int M_TOK = Bb * Ll;      // 4096 tokens
constexpr int HEADS = 8, HD = 128;
constexpr int VOCAB = 65536, TOOLV = 128;
constexpr float LNEPS = 1e-5f;

// A-fragment k mapping for 16-bit 16x32 A (contiguous: k = 16*g + e, matches B)
__device__ __forceinline__ int a_kmap(int e, int g) {
  return e + 16 * g;
}

// ---------------------------------------------------------------------------
// CDNA5 async global->LDS copy (16B per lane), tracked on ASYNCcnt.
// Generic pointers to __shared__ hold the LDS offset in their low 32 bits.
// ---------------------------------------------------------------------------
__device__ __forceinline__ unsigned lds_off(const void* p) {
  return (unsigned)(uintptr_t)p;
}
__device__ __forceinline__ void async_copy_b128(unsigned lds_byte, const void* gptr) {
  asm volatile("global_load_async_to_lds_b128 %0, %1, off"
               :: "v"(lds_byte), "v"(gptr) : "memory");
}
__device__ __forceinline__ void wait_async() {
  asm volatile("s_wait_asynccnt 0x0" ::: "memory");
}

// ---------------------------------------------------------------------------
// Julia-set feature embedding -> h (f32) and h16 (f16 mirror)
// ---------------------------------------------------------------------------
__global__ __launch_bounds__(256)
void k_embed(const float* __restrict__ cs, const float* __restrict__ pw,
             const float* __restrict__ esc, float* __restrict__ H,
             half_t* __restrict__ H16) {
  const int tok = blockIdx.x, tid = threadIdx.x;
  const float cr = cs[tok * 2 + 0], ci = cs[tok * 2 + 1];
  float f[16];
  float zr = 0.f, zi = 0.f;
#pragma unroll
  for (int s = 0; s < 8; ++s) {
    float nzr = zr * zr - zi * zi + cr;
    float nzi = 2.f * zr * zi + ci;
    zr = nzr; zi = nzi;
    f[2 * s] = zr; f[2 * s + 1] = zi;
  }
  const float e = esc[0];
#pragma unroll
  for (int j = 0; j < 4; ++j) {
    const int d = tid + 256 * j;
    float a = 0.f;
#pragma unroll
    for (int k = 0; k < 16; ++k) a += f[k] * pw[k * Dd + d];
    const float v = a * e;
    H[tok * Dd + d] = v;
    H16[tok * Dd + d] = (half_t)v;
  }
}

// ---------------------------------------------------------------------------
// Depthwise conv K=5 along L (zero pad) + bias; writes f16 GEMM operand.
// ---------------------------------------------------------------------------
__global__ __launch_bounds__(256)
void k_dwconv(const float* __restrict__ H, const float* __restrict__ w,
              const float* __restrict__ bias, half_t* __restrict__ Y) {
  const int id = blockIdx.x * 256 + threadIdx.x;
  const int d = id & (Dd - 1);
  const int tok = id >> 10;
  const int b = tok >> 11;
  const int l = tok & (Ll - 1);
  float acc = bias[d];
#pragma unroll
  for (int t = 0; t < 5; ++t) {
    const int ll = l + t - 2;
    if (ll >= 0 && ll < Ll)
      acc += H[(b * Ll + ll) * Dd + d] * w[d * 5 + t];
  }
  Y[id] = (half_t)acc;
}

// ---------------------------------------------------------------------------
// Weight conversion f32 -> f16 ([N][K] layout for contiguous B-fragments)
// ---------------------------------------------------------------------------
__global__ __launch_bounds__(256)
void k_cvt_copy(const float* __restrict__ in, half_t* __restrict__ out, int n) {
  const int id = blockIdx.x * 256 + threadIdx.x;
  if (id < n) out[id] = (half_t)in[id];
}

// LDS-tiled 32x32 transpose-convert: in [K][N] f32 -> out [N][K] f16.
__global__ __launch_bounds__(256)
void k_cvt_T(const float* __restrict__ in, half_t* __restrict__ out, int N, int K) {
  __shared__ float t[32][33];
  const int n0 = blockIdx.x * 32, k0 = blockIdx.y * 32;
  const int tid = threadIdx.x;
  const int r = tid >> 5, c = tid & 31;  // 8 rows x 32 cols per pass
#pragma unroll
  for (int j = 0; j < 4; ++j)
    t[r + 8 * j][c] = in[(k0 + r + 8 * j) * N + n0 + c];
  __syncthreads();
#pragma unroll
  for (int j = 0; j < 4; ++j)
    out[(n0 + r + 8 * j) * K + k0 + c] = (half_t)t[c][r + 8 * j];
}

// ---------------------------------------------------------------------------
// Tiled WMMA GEMM:  Y[M x N] = X[M x K](f16) * Wt^T,  Wt stored [N][K] f16.
// Workgroup tile 64(M) x 128(N); 8 wave32, each wave a 32x32 block
// (2x2 WMMA 16x16x32).  Both tiles async-staged, double-buffered.
// ---------------------------------------------------------------------------
__global__ __launch_bounds__(256)
void k_gemm(const half_t* __restrict__ X, const half_t* __restrict__ Wt,
            void* __restrict__ Yout, int K, int N,
            const float* __restrict__ bias, int out_f16) {
  constexpr int LD = 40;  // padded LDS row in halfs (80B -> 16B aligned rows)
  __shared__ half_t Xs[2][64 * LD];
  __shared__ half_t Ws[2][128 * LD];
  const int tid = threadIdx.x;
  const int lane = tid & 31, wave = tid >> 5;
  const int wm = (wave & 1) * 32;
  const int wn = (wave >> 1) * 32;
  const int mblk = blockIdx.y * 64, nblk = blockIdx.x * 128;
  const int lm = lane & 15, lg = lane >> 4;

  const unsigned xs0 = lds_off(&Xs[0][0]);
  const unsigned ws0 = lds_off(&Ws[0][0]);
  constexpr unsigned XBUF = 64 * LD * 2;   // bytes per X buffer
  constexpr unsigned WBUF = 128 * LD * 2;  // bytes per W buffer

  auto issue_tile = [&](int buf, int kb) {
    // X tile 64x32 f16: 256 x 16B chunks, 1 per thread
    {
      const int r = tid >> 2, sub = tid & 3;
      async_copy_b128(xs0 + buf * XBUF + r * (LD * 2) + sub * 16,
                      X + (mblk + r) * K + kb + sub * 8);
    }
    // W tile 128x32 f16: 512 x 16B chunks, 2 per thread
#pragma unroll
    for (int j = 0; j < 2; ++j) {
      const int cidx = tid + 256 * j;
      const int r = cidx >> 2, sub = cidx & 3;
      async_copy_b128(ws0 + buf * WBUF + r * (LD * 2) + sub * 16,
                      Wt + (nblk + r) * K + kb + sub * 8);
    }
  };

  v8f acc[2][2] = {};

  issue_tile(0, 0);
  wait_async();
  __syncthreads();

  const int nk = K / 32;
  for (int t = 0; t < nk; ++t) {
    const int p = t & 1;
    if (t + 1 < nk) issue_tile(1 - p, (t + 1) * 32);  // overlap with compute
    if (t + 2 < nk)  // gfx1250 global_prefetch_b8, two tiles ahead
      __builtin_prefetch(X + (mblk + (tid >> 2)) * K + (t + 2) * 32, 0, 1);

    v16h a[2];
#pragma unroll
    for (int tm = 0; tm < 2; ++tm) {
      const half_t* xr = &Xs[p][(wm + tm * 16 + lm) * LD];
#pragma unroll
      for (int e = 0; e < 16; ++e) a[tm][e] = xr[a_kmap(e, lg)];
    }
#pragma unroll
    for (int tn = 0; tn < 2; ++tn) {
      v16h bfr;
      const half_t* wr = &Ws[p][(wn + tn * 16 + lm) * LD + lg * 16];
#pragma unroll
      for (int e = 0; e < 16; ++e) bfr[e] = wr[e];
      acc[0][tn] = WMMA_F32_F16(a[0], bfr, acc[0][tn]);
      acc[1][tn] = WMMA_F32_F16(a[1], bfr, acc[1][tn]);
    }
    wait_async();
    __syncthreads();
  }

  // Epilogue: C layout — VGPR r: m = r + 8*(lane>>4); n = lane&15.
#pragma unroll
  for (int tm = 0; tm < 2; ++tm)
#pragma unroll
    for (int tn = 0; tn < 2; ++tn)
#pragma unroll
      for (int r = 0; r < 8; ++r) {
        const int m = mblk + wm + tm * 16 + lg * 8 + r;
        const int n = nblk + wn + tn * 16 + lm;
        float v = acc[tm][tn][r];
        if (bias) v += bias[n];
        if (out_f16) ((half_t*)Yout)[m * N + n] = (half_t)v;
        else         ((float*)Yout)[m * N + n] = v;
      }
}

// ---------------------------------------------------------------------------
// Flash attention: 4 wave32 per block, 64 q rows (16/wave), one (batch,head).
// K/V tiles async-staged and double-buffered.  16 WMMA per key tile.
// Context written as f16 for the O-projection GEMM.
// ---------------------------------------------------------------------------
__global__ __launch_bounds__(128)
void k_flash(const half_t* __restrict__ Q, const half_t* __restrict__ KV,
             half_t* __restrict__ O16, float scale) {
  __shared__ half_t Qs[64 * 128];
  __shared__ half_t Ks[2][32 * 128];
  __shared__ half_t Vs[2][32 * 128];
  __shared__ half_t Ps[4][16 * 32];

  const int tid = threadIdx.x, lane = tid & 31, wave = tid >> 5;
  const int lm = lane & 15, lg = lane >> 4;
  const int bh = blockIdx.y;
  const int b = bh >> 3, hh = bh & 7;
  const int q0 = blockIdx.x * 64;

  const unsigned qs_base = lds_off(&Qs[0]);
  const unsigned ks_base = lds_off(&Ks[0][0]);
  const unsigned vs_base = lds_off(&Vs[0][0]);
  constexpr unsigned KVBUF = 32 * 128 * 2;  // bytes per K (or V) buffer

  auto issue_kv = [&](int buf, int it) {
#pragma unroll
    for (int j = 0; j < 4; ++j) {
      const int cidx = tid + 128 * j;           // 0..511 chunks of 16B
      const int r = cidx >> 4, sub = cidx & 15;
      const int grow = (b * Ll + it * 32 + r) * (2 * Dd) + hh * 256;
      async_copy_b128(ks_base + buf * KVBUF + cidx * 16, KV + grow + sub * 8);
      async_copy_b128(vs_base + buf * KVBUF + cidx * 16, KV + grow + 128 + sub * 8);
    }
  };

  // Stage Q tile 64x128 f16 (async) + first K/V tile
#pragma unroll
  for (int j = 0; j < 8; ++j) {
    const int cidx = tid + 128 * j;           // 0..1023
    const int r = cidx >> 4, sub = cidx & 15;
    async_copy_b128(qs_base + cidx * 16,
                    Q + (b * Ll + q0 + r) * Dd + hh * HD + sub * 8);
  }
  issue_kv(0, 0);
  wait_async();
  __syncthreads();

  v16h qa[4];
#pragma unroll
  for (int kc = 0; kc < 4; ++kc) {
    const half_t* qr = &Qs[(wave * 16 + lm) * 128 + kc * 32];
#pragma unroll
    for (int e = 0; e < 16; ++e) qa[kc][e] = qr[a_kmap(e, lg)];
  }

  float Mx[8], Ls[8];
  v8f acc[8] = {};
#pragma unroll
  for (int r = 0; r < 8; ++r) { Mx[r] = -1e30f; Ls[r] = 0.f; }

  constexpr int NIT = Ll / 32;
  for (int it = 0; it < NIT; ++it) {
    const int p = it & 1;
    if (it + 1 < NIT) issue_kv(1 - p, it + 1);  // overlap with compute

    // S = Q Kt : two 16x16 tiles per wave, 4 WMMA each over HD=128
    v8f s0 = {}, s1 = {};
#pragma unroll
    for (int kc = 0; kc < 4; ++kc) {
      v16h bk0, bk1;
      const half_t* k0 = &Ks[p][lm * 128 + kc * 32 + lg * 16];
      const half_t* k1 = &Ks[p][(16 + lm) * 128 + kc * 32 + lg * 16];
#pragma unroll
      for (int e = 0; e < 16; ++e) { bk0[e] = k0[e]; bk1[e] = k1[e]; }
      s0 = WMMA_F32_F16(qa[kc], bk0, s0);
      s1 = WMMA_F32_F16(qa[kc], bk1, s1);
    }

    float al[8];
#pragma unroll
    for (int r = 0; r < 8; ++r) {
      const float v0 = s0[r] * scale, v1 = s1[r] * scale;
      float mx = fmaxf(v0, v1);
#pragma unroll
      for (int m = 1; m < 16; m <<= 1) mx = fmaxf(mx, __shfl_xor(mx, m, 32));
      const float mn = fmaxf(Mx[r], mx);
      al[r] = __expf(Mx[r] - mn);
      const float e0 = __expf(v0 - mn), e1 = __expf(v1 - mn);
      float rs = e0 + e1;
#pragma unroll
      for (int m = 1; m < 16; m <<= 1) rs += __shfl_xor(rs, m, 32);
      Ls[r] = Ls[r] * al[r] + rs;
      Mx[r] = mn;
      Ps[wave][(r + lg * 8) * 32 + lm]      = (half_t)e0;
      Ps[wave][(r + lg * 8) * 32 + 16 + lm] = (half_t)e1;
    }

#pragma unroll
    for (int j = 0; j < 8; ++j)
#pragma unroll
      for (int r = 0; r < 8; ++r) acc[j][r] *= al[r];

    v16h pa;
    const half_t* pr = &Ps[wave][lm * 32];
#pragma unroll
    for (int e = 0; e < 16; ++e) pa[e] = pr[a_kmap(e, lg)];

#pragma unroll
    for (int j = 0; j < 8; ++j) {
      v16h bv;
#pragma unroll
      for (int e = 0; e < 16; ++e) bv[e] = Vs[p][(lg * 16 + e) * 128 + j * 16 + lm];
      acc[j] = WMMA_F32_F16(pa, bv, acc[j]);
    }
    wait_async();
    __syncthreads();
  }

#pragma unroll
  for (int j = 0; j < 8; ++j)
#pragma unroll
    for (int r = 0; r < 8; ++r) {
      const int m = wave * 16 + lg * 8 + r;
      O16[(b * Ll + q0 + m) * Dd + hh * HD + j * 16 + lm] =
          (half_t)(acc[j][r] / Ls[r]);
    }
}

// ---------------------------------------------------------------------------
// Fused residual (+GLU) + LayerNorm.  Updates f32 stream and f16 mirror.
// ---------------------------------------------------------------------------
__global__ __launch_bounds__(256)
void k_fuse_ln(float* __restrict__ H, half_t* __restrict__ H16,
               const float* __restrict__ Y,
               const float* __restrict__ s, const float* __restrict__ bvec,
               int mode) {
  __shared__ float red[256];
  const int tok = blockIdx.x, tid = threadIdx.x;
  float rr[4];
  float sum = 0.f;
#pragma unroll
  for (int j = 0; j < 4; ++j) {
    const int d = tid + 256 * j;
    const float x = H[tok * Dd + d];
    float v;
    if (mode == 0) {
      const float a = Y[tok * 2048 + d];
      const float g = Y[tok * 2048 + 1024 + d];
      v = x + a * (1.f / (1.f + __expf(-g)));
    } else {
      v = x + Y[tok * Dd + d];
    }
    rr[j] = v; sum += v;
  }
  red[tid] = sum; __syncthreads();
  for (int st = 128; st > 0; st >>= 1) { if (tid < st) red[tid] += red[tid + st]; __syncthreads(); }
  const float mu = red[0] / Dd;
  __syncthreads();
  float vs = 0.f;
#pragma unroll
  for (int j = 0; j < 4; ++j) { const float df = rr[j] - mu; vs += df * df; }
  red[tid] = vs; __syncthreads();
  for (int st = 128; st > 0; st >>= 1) { if (tid < st) red[tid] += red[tid + st]; __syncthreads(); }
  const float inv = rsqrtf(red[0] / Dd + LNEPS);
#pragma unroll
  for (int j = 0; j < 4; ++j) {
    const int d = tid + 256 * j;
    const float o = (rr[j] - mu) * inv * s[d] + bvec[d];
    H[tok * Dd + d] = o;
    H16[tok * Dd + d] = (half_t)o;
  }
}

// ---------------------------------------------------------------------------
// Heads
// ---------------------------------------------------------------------------
__global__ __launch_bounds__(256)
void k_lmhead(const float* __restrict__ H, const float* __restrict__ W,
              float* __restrict__ out) {
  const int col = blockIdx.x * 256 + threadIdx.x;
  const float* h0 = &H[(Ll - 1) * Dd];
  const float* h1 = &H[(2 * Ll - 1) * Dd];
  float a0 = 0.f, a1 = 0.f;
#pragma unroll 4
  for (int k = 0; k < Dd; ++k) {
    const float w = W[k * VOCAB + col];
    a0 += h0[k] * w; a1 += h1[k] * w;
  }
  out[col] = a0;
  out[VOCAB + col] = a1;
}

__global__ __launch_bounds__(128)
void k_toolhead(const float* __restrict__ H, const float* __restrict__ W,
                const float* __restrict__ bias, float* __restrict__ out) {
  const int col = threadIdx.x;
  const float* h0 = &H[(Ll - 1) * Dd];
  const float* h1 = &H[(2 * Ll - 1) * Dd];
  float a0 = 0.f, a1 = 0.f;
#pragma unroll 4
  for (int k = 0; k < Dd; ++k) {
    const float w = W[k * TOOLV + col];
    a0 += h0[k] * w; a1 += h1[k] * w;
  }
  out[2 * VOCAB + col] = a0 + bias[col];
  out[2 * VOCAB + TOOLV + col] = a1 + bias[col];
}

// ---------------------------------------------------------------------------
// Host orchestration
// ---------------------------------------------------------------------------
extern "C" void kernel_launch(void* const* d_in, const int* in_sizes, int n_in,
                              void* d_out, int out_size, void* d_ws, size_t ws_size,
                              hipStream_t stream) {
  (void)in_sizes; (void)n_in; (void)out_size; (void)ws_size;
  const float* cs     = (const float*)d_in[1];
  const float* proj_w = (const float*)d_in[2];
  const float* emb_sc = (const float*)d_in[3];
  const float* cdw_w  = (const float*)d_in[4];
  const float* cdw_b  = (const float*)d_in[5];
  const float* cpw_w  = (const float*)d_in[6];
  const float* cpw_b  = (const float*)d_in[7];
  const float* cln_s  = (const float*)d_in[8];
  const float* cln_b  = (const float*)d_in[9];
  const float* aq_w   = (const float*)d_in[10];
  const float* akv_w  = (const float*)d_in[11];
  const float* ao_w   = (const float*)d_in[12];
  const float* aln_s  = (const float*)d_in[13];
  const float* aln_b  = (const float*)d_in[14];
  const float* lm_w   = (const float*)d_in[15];
  const float* tool_w = (const float*)d_in[16];
  const float* tool_b = (const float*)d_in[17];
  float* out = (float*)d_out;

  char* ws = (char*)d_ws;
  size_t off = 0;
  auto carve = [&](size_t bytes) {
    char* p = ws + off;
    off += (bytes + 255) & ~(size_t)255;
    return p;
  };
  float*  h      = (float*) carve((size_t)M_TOK * Dd * 4);       // residual f32
  float*  y2     = (float*) carve((size_t)M_TOK * 2 * Dd * 4);   // GEMM out f32
  half_t* h16    = (half_t*)carve((size_t)M_TOK * Dd * 2);       // residual f16
  half_t* xf16   = (half_t*)carve((size_t)M_TOK * Dd * 2);       // dwconv out f16
  half_t* wstage = (half_t*)carve((size_t)2 * Dd * Dd * 2);      // weights f16
  half_t* qbuf   = (half_t*)carve((size_t)M_TOK * Dd * 2);       // q f16
  half_t* kvbuf  = (half_t*)carve((size_t)M_TOK * 2 * Dd * 2);   // kv f16
  half_t* ctx16  = (half_t*)carve((size_t)M_TOK * Dd * 2);       // attn ctx f16

  const float scale = 0.08838834764831845f;  // 128^-0.5

  k_embed<<<M_TOK, 256, 0, stream>>>(cs, proj_w, emb_sc, h, h16);

  for (int i = 0; i < 10; ++i) {
    k_dwconv<<<(M_TOK * Dd) / 256, 256, 0, stream>>>(
        h, cdw_w + (size_t)i * Dd * 5, cdw_b + (size_t)i * Dd, xf16);
    const int npw = 2 * Dd * Dd;
    k_cvt_copy<<<npw / 256, 256, 0, stream>>>(cpw_w + (size_t)i * npw, wstage, npw);
    k_gemm<<<dim3(2 * Dd / 128, M_TOK / 64), 256, 0, stream>>>(
        xf16, wstage, y2, Dd, 2 * Dd, cpw_b + (size_t)i * 2 * Dd, 0);
    k_fuse_ln<<<M_TOK, 256, 0, stream>>>(h, h16, y2, cln_s + (size_t)i * Dd,
                                         cln_b + (size_t)i * Dd, 0);
  }

  for (int i = 0; i < 6; ++i) {
    k_cvt_T<<<dim3(Dd / 32, Dd / 32), 256, 0, stream>>>(
        aq_w + (size_t)i * Dd * Dd, wstage, Dd, Dd);
    k_gemm<<<dim3(Dd / 128, M_TOK / 64), 256, 0, stream>>>(
        h16, wstage, qbuf, Dd, Dd, nullptr, 1);
    k_cvt_T<<<dim3(2 * Dd / 32, Dd / 32), 256, 0, stream>>>(
        akv_w + (size_t)i * 2 * Dd * Dd, wstage, 2 * Dd, Dd);
    k_gemm<<<dim3(2 * Dd / 128, M_TOK / 64), 256, 0, stream>>>(
        h16, wstage, kvbuf, Dd, 2 * Dd, nullptr, 1);
    k_flash<<<dim3(Ll / 64, Bb * HEADS), 128, 0, stream>>>(qbuf, kvbuf, ctx16, scale);
    k_cvt_T<<<dim3(Dd / 32, Dd / 32), 256, 0, stream>>>(
        ao_w + (size_t)i * Dd * Dd, wstage, Dd, Dd);
    k_gemm<<<dim3(Dd / 128, M_TOK / 64), 256, 0, stream>>>(
        ctx16, wstage, y2, Dd, Dd, nullptr, 0);
    k_fuse_ln<<<M_TOK, 256, 0, stream>>>(h, h16, y2, aln_s + (size_t)i * Dd,
                                         aln_b + (size_t)i * Dd, 1);
  }

  k_lmhead<<<VOCAB / 256, 256, 0, stream>>>(h, lm_w, out);
  k_toolhead<<<1, TOOLV, 0, stream>>>(h, tool_w, tool_b, out);
}
